// DeformableTransformerLayerForVisualEncoder_10874857193577
// MI455X (gfx1250) — matrix operand order repaired and checked by reading, not compiled, not run on hardware
//
#include <hip/hip_runtime.h>

// ---------------------------------------------------------------------------
// Problem constants (fixed in the reference)
// ---------------------------------------------------------------------------
constexpr int Bq  = 8;
constexpr int LQ  = 1024;
constexpr int Dm  = 256;
constexpr int FF  = 1024;
constexpr int NH  = 8;
constexpr int NL  = 4;
constexpr int NP  = 4;
constexpr int HD  = Dm / NH;          // 32
constexpr int LIN = 13294;
constexpr int MQ  = Bq * LQ;          // 8192 rows
constexpr int MS  = Bq * LIN;         // 106352 rows (divisible by 16)

__constant__ __device__ const int cH[NL] = {100, 50, 25, 13};
__constant__ __device__ const int cW[NL] = {100, 50, 25, 13};
__constant__ __device__ const int cS[NL] = {0, 10000, 12500, 13125};

typedef __attribute__((ext_vector_type(16))) _Float16 v16h;
typedef __attribute__((ext_vector_type(8)))  _Float16 v8h;
typedef __attribute__((ext_vector_type(8)))  float    v8f;
typedef __attribute__((ext_vector_type(4)))  unsigned int v4u;
typedef __attribute__((ext_vector_type(8)))  int          v8i_;
typedef __attribute__((ext_vector_type(4)))  int          v4i_;

// ---------------------------------------------------------------------------
// WMMA fragment helpers (wave32, 16x16x32 f16 -> f32)
//  A-frag (16x32, M x K): lane l -> row = l%16; K halves: (l/16)*8 .. +7 and
//                         16+(l/16)*8 .. +7  (two 16B loads)
//  B-frag (W stored [N,K]): lane l -> n = l%16; K = (l/16)*8 .. +7 and 16+(l/16)*8 .. +7
//  C-frag: lane l -> col n = l%16, rows m = r + 8*(l/16), r = 0..7
// ---------------------------------------------------------------------------
__device__ __forceinline__ v16h cat8(v8h lo, v8h hi) {
  v16h r;
#pragma unroll
  for (int i = 0; i < 8; ++i) { r[i] = lo[i]; r[8 + i] = hi[i]; }
  return r;
}

__device__ __forceinline__ v16h load_a_frag(const _Float16* A, int lda,
                                            int row0, int k0, int lane) {
  int m = lane & 15, sub = lane >> 4;
  const _Float16* p = A + (size_t)(row0 + m) * lda + k0 + sub * 8;
  v8h lo = *(const v8h*)p;
  v8h hi = *(const v8h*)(p + 16);
  return cat8(lo, hi);
}

__device__ __forceinline__ v16h load_b_frag(const _Float16* W, int ldw,
                                            int n0, int k0, int lane) {
  int n = lane & 15, sub = lane >> 4;
  const _Float16* p = W + (size_t)(n0 + n) * ldw + k0 + sub * 8;
  v8h lo = *(const v8h*)p;
  v8h hi = *(const v8h*)(p + 16);
  return cat8(lo, hi);
}

__device__ __forceinline__ v8f wmma_f16(v16h a, v16h b, v8f c) {
  return __builtin_amdgcn_wmma_f32_16x16x32_f16(false, a, false, b,
                                                (short)0, c, false, false);
}

__device__ __forceinline__ void store_tile(float* C, int ldc, int row0, int n0,
                                           const v8f* accs, const float* bias,
                                           int relu, int lane) {
  int n = lane & 15, sub = lane >> 4;
#pragma unroll
  for (int t = 0; t < 4; ++t) {
    int col = n0 + t * 16 + n;
    float bv = bias ? bias[col] : 0.f;
#pragma unroll
    for (int r = 0; r < 8; ++r) {
      int m = row0 + r + 8 * sub;
      float v = accs[t][r] + bv;
      if (relu) v = fmaxf(v, 0.f);
      C[(size_t)m * ldc + col] = v;
    }
  }
}

// ---------------------------------------------------------------------------
// Generic GEMM:  C[M,N] = A[M,K](f16) @ W[N,K](f16)^T + bias, opt ReLU.
// One wave per block computes a 16x64 tile; fragments double-buffered so the
// next K-step's VMEM overlaps the current WMMAs.  grid = (M/16, N/64)
// ---------------------------------------------------------------------------
__global__ void k_gemm(const _Float16* __restrict__ A, int lda,
                       const _Float16* __restrict__ W, int ldw,
                       const float* __restrict__ bias,
                       float* __restrict__ C, int ldc,
                       int K, int relu) {
  int lane = threadIdx.x;
  int row0 = blockIdx.x * 16;
  int n0   = blockIdx.y * 64;

  v8f acc[4] = {};
  v16h a_cur, b_cur[4];
  a_cur = load_a_frag(A, lda, row0, 0, lane);
#pragma unroll
  for (int t = 0; t < 4; ++t) b_cur[t] = load_b_frag(W, ldw, n0 + t * 16, 0, lane);

  for (int k0 = 0; k0 < K; k0 += 32) {
    v16h a_nxt = a_cur, b_nxt[4];
#pragma unroll
    for (int t = 0; t < 4; ++t) b_nxt[t] = b_cur[t];
    int kn = k0 + 32;
    if (kn < K) {
      __builtin_prefetch(A + (size_t)row0 * lda + kn + 32, 0, 1);
      a_nxt = load_a_frag(A, lda, row0, kn, lane);
#pragma unroll
      for (int t = 0; t < 4; ++t) b_nxt[t] = load_b_frag(W, ldw, n0 + t * 16, kn, lane);
    }
#pragma unroll
    for (int t = 0; t < 4; ++t) acc[t] = wmma_f16(a_cur, b_cur[t], acc[t]);
    a_cur = a_nxt;
#pragma unroll
    for (int t = 0; t < 4; ++t) b_cur[t] = b_nxt[t];
  }
  store_tile(C, ldc, row0, n0, acc, bias, relu, lane);
}

// ---------------------------------------------------------------------------
// K=256 GEMM with Tensor Data Mover: the 16x256 f16 A-tile is DMA'd to LDS by
// one TENSOR_LOAD_TO_LDS (D# built per ISA 8.3/8.4; 2D tile, data_size=2B),
// waited with s_wait_tensorcnt, then A-frags come from ds_load_b128 while the
// B-frags are double-buffered from global.
// ---------------------------------------------------------------------------
__global__ void k_gemm_tdm(const _Float16* __restrict__ A, int lda,
                           const _Float16* __restrict__ W, int ldw,
                           const float* __restrict__ bias,
                           float* __restrict__ C, int ldc, int relu) {
  constexpr int K = 256;
  __shared__ __align__(16) _Float16 As[16 * K];
  int lane = threadIdx.x;
  int row0 = blockIdx.x * 16;
  int n0   = blockIdx.y * 64;

  // ---- Tensor DMA descriptor (all operands uniform -> SGPR groups) ----
  unsigned long long gaddr = (unsigned long long)(const void*)(A + (size_t)row0 * lda);
  unsigned int ldsa = (unsigned int)(unsigned long long)(__SIZE_TYPE__)(void*)As;
  v4u g0;
  g0[0] = 1u;                                              // count=1, user mode
  g0[1] = ldsa;                                            // lds_addr
  g0[2] = (unsigned int)(gaddr & 0xffffffffull);           // global_addr[31:0]
  g0[3] = (unsigned int)((gaddr >> 32) & 0x01ffffffull)    // global_addr[56:32]
          | (2u << 30);                                    // type=2 (image)
  unsigned int dim0 = (unsigned)K;                         // tensor_dim0 (elems)
  unsigned int dim1 = 16u;                                 // tensor_dim1 (rows)
  unsigned int str0 = (unsigned)lda;                       // dim0 stride (elems)
  v8i_ g1;
  g1[0] = (int)(1u << 16);                                 // data_size=1 (2B)
  g1[1] = (int)((dim0 & 0xffffu) << 16);                   // tensor_dim0 lo
  g1[2] = (int)(((dim0 >> 16) & 0xffffu) | ((dim1 & 0xffffu) << 16));
  g1[3] = (int)(((dim1 >> 16) & 0xffffu) | ((unsigned)K << 16)); // tile_dim0=K
  g1[4] = (int)16u;                                        // tile_dim1=16, tile_dim2=0
  g1[5] = (int)str0;                                       // dim0_stride lo
  g1[6] = 0;                                               // stride hi, dim1_stride lo
  g1[7] = 0;
  v4i_ gz4 = {0, 0, 0, 0};
  v8i_ gz8 = {0, 0, 0, 0, 0, 0, 0, 0};
  __builtin_amdgcn_tensor_load_to_lds(g0, g1, gz4, gz4, gz8, 0);
  __builtin_amdgcn_s_wait_tensorcnt((short)0);

  v8f acc[4] = {};
  v16h b_cur[4];
#pragma unroll
  for (int t = 0; t < 4; ++t) b_cur[t] = load_b_frag(W, ldw, n0 + t * 16, 0, lane);

  for (int k0 = 0; k0 < K; k0 += 32) {
    v16h b_nxt[4];
#pragma unroll
    for (int t = 0; t < 4; ++t) b_nxt[t] = b_cur[t];
    int kn = k0 + 32;
    if (kn < K) {
#pragma unroll
      for (int t = 0; t < 4; ++t) b_nxt[t] = load_b_frag(W, ldw, n0 + t * 16, kn, lane);
    }
    v16h a = load_a_frag(As, K, 0, k0, lane);              // ds_load_b128 x2
#pragma unroll
    for (int t = 0; t < 4; ++t) acc[t] = wmma_f16(a, b_cur[t], acc[t]);
#pragma unroll
    for (int t = 0; t < 4; ++t) b_cur[t] = b_nxt[t];
  }
  store_tile(C, ldc, row0, n0, acc, bias, relu, lane);
}

// ---------------------------------------------------------------------------
// Elementwise helpers
// ---------------------------------------------------------------------------
__global__ void k_add_cvt(const float* __restrict__ x, const float* __restrict__ y,
                          _Float16* __restrict__ out, long long n) {
  long long i = blockIdx.x * (long long)blockDim.x + threadIdx.x;
  if (i < n) out[i] = (_Float16)(x[i] + y[i]);
}

__global__ void k_cvt(const float* __restrict__ x, _Float16* __restrict__ out,
                      long long n) {
  long long i = blockIdx.x * (long long)blockDim.x + threadIdx.x;
  if (i < n) out[i] = (_Float16)x[i];
}

// ---------------------------------------------------------------------------
// MSDA bilinear sampling.  One wave per (b,q,h); lane = channel (HD=32).
// ---------------------------------------------------------------------------
__global__ void k_msda(const float* __restrict__ off,
                       const float* __restrict__ logit,
                       const float* __restrict__ ref,
                       const float* __restrict__ val,
                       float* __restrict__ acc, _Float16* __restrict__ acch) {
  int wid = (int)((blockIdx.x * (long long)blockDim.x + threadIdx.x) >> 5);
  int c   = threadIdx.x & 31;
  if (wid >= MQ * NH) return;
  int h  = wid % NH;
  int bq = wid / NH;          // b*LQ + q
  int b  = bq / LQ;

  const float* lg = logit + (size_t)bq * (NH * NL * NP) + h * 16;
  float mx = -1e30f;
#pragma unroll
  for (int i = 0; i < 16; ++i) mx = fmaxf(mx, lg[i]);
  float e[16], s = 0.f;
#pragma unroll
  for (int i = 0; i < 16; ++i) { e[i] = __expf(lg[i] - mx); s += e[i]; }
  float inv = 1.f / s;

  const float* of = off + (size_t)bq * Dm + h * (NL * NP * 2);
  float a = 0.f;
#pragma unroll
  for (int l = 0; l < NL; ++l) {
    int H = cH[l], W = cW[l], st = cS[l];
    float rx = ref[((size_t)bq * NL + l) * 2 + 0];
    float ry = ref[((size_t)bq * NL + l) * 2 + 1];
    const float* vbase = val + ((size_t)b * LIN + st) * Dm + h * HD + c;
#pragma unroll
    for (int p = 0; p < NP; ++p) {
      float ox = of[l * 8 + p * 2 + 0];
      float oy = of[l * 8 + p * 2 + 1];
      float x = rx * (float)W + ox - 0.5f;     // (rx + ox/W)*W - 0.5
      float y = ry * (float)H + oy - 0.5f;
      float x0f = floorf(x), y0f = floorf(y);
      float wx = x - x0f, wy = y - y0f;
      int x0 = (int)x0f, y0 = (int)y0f;
      auto g = [&](int iy, int ix) -> float {
        bool valid = (iy >= 0) && (iy < H) && (ix >= 0) && (ix < W);
        int cy = iy < 0 ? 0 : (iy >= H ? H - 1 : iy);
        int cx = ix < 0 ? 0 : (ix >= W ? W - 1 : ix);
        float vv = vbase[(size_t)(cy * W + cx) * Dm];
        return valid ? vv : 0.f;
      };
      float samp = g(y0, x0)         * (1.f - wy) * (1.f - wx)
                 + g(y0, x0 + 1)     * (1.f - wy) * wx
                 + g(y0 + 1, x0)     * wy * (1.f - wx)
                 + g(y0 + 1, x0 + 1) * wy * wx;
      a += (e[l * 4 + p] * inv) * samp;
    }
  }
  size_t o = (size_t)bq * Dm + h * HD + c;
  acc[o]  = a;
  acch[o] = (_Float16)a;
}

// ---------------------------------------------------------------------------
// Residual + LayerNorm over rows of 256.  One wave per row (8 elems/lane).
// ---------------------------------------------------------------------------
__global__ void k_ln_res(const float* __restrict__ x, const float* __restrict__ res,
                         const float* __restrict__ g, const float* __restrict__ bta,
                         float* __restrict__ out, _Float16* __restrict__ outh,
                         int rows) {
  int wid  = (int)((blockIdx.x * (long long)blockDim.x + threadIdx.x) >> 5);
  int lane = threadIdx.x & 31;
  if (wid >= rows) return;
  const float* xr = x   + (size_t)wid * Dm;
  const float* rr = res + (size_t)wid * Dm;
  float v[8], s = 0.f;
#pragma unroll
  for (int i = 0; i < 8; ++i) { v[i] = xr[lane + i * 32] + rr[lane + i * 32]; s += v[i]; }
  for (int o = 16; o >= 1; o >>= 1) s += __shfl_xor(s, o, 32);
  float mean = s * (1.f / 256.f);
  float vs = 0.f;
#pragma unroll
  for (int i = 0; i < 8; ++i) { float d = v[i] - mean; vs += d * d; }
  for (int o = 16; o >= 1; o >>= 1) vs += __shfl_xor(vs, o, 32);
  float rstd = rsqrtf(vs * (1.f / 256.f) + 1e-5f);
#pragma unroll
  for (int i = 0; i < 8; ++i) {
    int cix = lane + i * 32;
    float y = (v[i] - mean) * rstd * g[cix] + bta[cix];
    out[(size_t)wid * Dm + cix] = y;
    if (outh) outh[(size_t)wid * Dm + cix] = (_Float16)y;
  }
}

// ---------------------------------------------------------------------------
// Split fused QKV into per-head f16 Q (pre-scaled by 1/sqrt(HD)), K, and V^T.
// ---------------------------------------------------------------------------
__global__ void k_qkv_split(const float* __restrict__ qkv,
                            _Float16* __restrict__ qh, _Float16* __restrict__ kh,
                            _Float16* __restrict__ vT) {
  long long i = blockIdx.x * (long long)blockDim.x + threadIdx.x;
  if (i >= (long long)MQ * Dm) return;
  int ch = (int)(i & (Dm - 1));
  long long bq = i >> 8;
  int q = (int)(bq % LQ);
  int b = (int)(bq / LQ);
  int h = ch >> 5, c = ch & 31;
  const float* base = qkv + (size_t)bq * (3 * Dm);
  size_t hq = ((size_t)b * NH + h) * LQ + q;
  qh[hq * HD + c] = (_Float16)(base[ch] * 0.17677669529663687f);
  kh[hq * HD + c] = (_Float16)(base[Dm + ch]);
  vT[(((size_t)b * NH + h) * HD + c) * LQ + q] = (_Float16)(base[2 * Dm + ch]);
}

// ---------------------------------------------------------------------------
// Flash self-attention: one wave per (b*NH, 16-query tile).
//   S tile  : 1 WMMA per 16-key tile (K = HD = 32 exactly)
//   P @ V   : P transposed C-frag -> A-frag through LDS, V^T rows as B-frags
// Online softmax kept in registers (8 rows/lane, row group = 16 lanes).
// ---------------------------------------------------------------------------
__global__ void k_flash(const _Float16* __restrict__ qh,
                        const _Float16* __restrict__ kh,
                        const _Float16* __restrict__ vT,
                        float* __restrict__ o, _Float16* __restrict__ oh) {
  __shared__ __align__(16) _Float16 pls[16 * 32];
  int lane = threadIdx.x;
  int qt = blockIdx.x, bh = blockIdx.y;
  const _Float16* Q  = qh + (size_t)bh * LQ * HD;
  const _Float16* Kp = kh + (size_t)bh * LQ * HD;
  const _Float16* V  = vT + (size_t)bh * HD * LQ;

  v16h qa = load_a_frag(Q, HD, qt * 16, 0, lane);
  float m_[8], l_[8];
  v8f o0 = {}, o1 = {};
#pragma unroll
  for (int r = 0; r < 8; ++r) { m_[r] = -1e30f; l_[r] = 0.f; }
  int n = lane & 15, sub = lane >> 4;

  for (int k0 = 0; k0 < LQ; k0 += 32) {
    v16h bk0 = load_b_frag(Kp, HD, k0, 0, lane);
    v16h bk1 = load_b_frag(Kp, HD, k0 + 16, 0, lane);
    v8f s0 = {}, s1 = {};
    s0 = wmma_f16(qa, bk0, s0);
    s1 = wmma_f16(qa, bk1, s1);
#pragma unroll
    for (int r = 0; r < 8; ++r) {
      float mx = fmaxf(s0[r], s1[r]);
      for (int ofs = 8; ofs >= 1; ofs >>= 1) mx = fmaxf(mx, __shfl_xor(mx, ofs, 32));
      float mn = fmaxf(m_[r], mx);
      float sc = __expf(m_[r] - mn);
      float p0 = __expf(s0[r] - mn);
      float p1 = __expf(s1[r] - mn);
      float ls = p0 + p1;
      for (int ofs = 8; ofs >= 1; ofs >>= 1) ls += __shfl_xor(ls, ofs, 32);
      l_[r] = l_[r] * sc + ls;
      m_[r] = mn;
      o0[r] *= sc; o1[r] *= sc;
      pls[(r + 8 * sub) * 32 + n]      = (_Float16)p0;
      pls[(r + 8 * sub) * 32 + 16 + n] = (_Float16)p1;
    }
    __syncthreads();
    v16h pa = load_a_frag(pls, 32, 0, 0, lane);
    __syncthreads();
    v16h bv0 = load_b_frag(V, LQ,  0, k0, lane);   // hd rows 0..15
    v16h bv1 = load_b_frag(V, LQ, 16, k0, lane);   // hd rows 16..31
    o0 = wmma_f16(pa, bv0, o0);
    o1 = wmma_f16(pa, bv1, o1);
  }

  int h = bh % NH, b = bh / NH;
#pragma unroll
  for (int r = 0; r < 8; ++r) {
    int qrow = qt * 16 + r + 8 * sub;
    float inv = 1.f / l_[r];
    size_t base = ((size_t)b * LQ + qrow) * Dm + h * HD;
    float v0 = o0[r] * inv, v1 = o1[r] * inv;
    o[base + n]       = v0;
    o[base + 16 + n]  = v1;
    oh[base + n]      = (_Float16)v0;
    oh[base + 16 + n] = (_Float16)v1;
  }
}

// ---------------------------------------------------------------------------
// Host orchestration
// ---------------------------------------------------------------------------
extern "C" void kernel_launch(void* const* d_in, const int* in_sizes, int n_in,
                              void* d_out, int out_size, void* d_ws, size_t ws_size,
                              hipStream_t stream) {
  const float* tgt   = (const float*)d_in[0];
  const float* qpos  = (const float*)d_in[1];
  const float* ref   = (const float*)d_in[2];
  const float* src   = (const float*)d_in[3];
  const float* W_off = (const float*)d_in[6];
  const float* b_off = (const float*)d_in[7];
  const float* W_att = (const float*)d_in[8];
  const float* b_att = (const float*)d_in[9];
  const float* W_val = (const float*)d_in[10];
  const float* b_val = (const float*)d_in[11];
  const float* W_out = (const float*)d_in[12];
  const float* b_out = (const float*)d_in[13];
  const float* ln1g  = (const float*)d_in[14];
  const float* ln1b  = (const float*)d_in[15];
  const float* Wproj = (const float*)d_in[16];
  const float* bproj = (const float*)d_in[17];
  const float* Wmha  = (const float*)d_in[18];
  const float* bmha  = (const float*)d_in[19];
  const float* ln2g  = (const float*)d_in[20];
  const float* ln2b  = (const float*)d_in[21];
  const float* W1    = (const float*)d_in[22];
  const float* b1    = (const float*)d_in[23];
  const float* W2    = (const float*)d_in[24];
  const float* b2    = (const float*)d_in[25];
  const float* ln3g  = (const float*)d_in[26];
  const float* ln3b  = (const float*)d_in[27];
  float* out = (float*)d_out;

  char* w = (char*)d_ws;
  size_t off_b = 0;
  auto take = [&](size_t bytes) -> void* {
    void* p = w + off_b;
    off_b = (off_b + bytes + 255) & ~(size_t)255;
    return p;
  };

  _Float16* qh    = (_Float16*)take((size_t)MQ * Dm * 2);
  _Float16* srch  = (_Float16*)take((size_t)MS * Dm * 2);
  _Float16* whoff = (_Float16*)take((size_t)256 * 256 * 2);
  _Float16* whatt = (_Float16*)take((size_t)128 * 256 * 2);
  _Float16* whval = (_Float16*)take((size_t)256 * 256 * 2);
  _Float16* whout = (_Float16*)take((size_t)256 * 256 * 2);
  _Float16* whprj = (_Float16*)take((size_t)768 * 256 * 2);
  _Float16* whmha = (_Float16*)take((size_t)256 * 256 * 2);
  _Float16* whw1  = (_Float16*)take((size_t)1024 * 256 * 2);
  _Float16* whw2  = (_Float16*)take((size_t)256 * 1024 * 2);
  float*    offp  = (float*)take((size_t)MQ * Dm * 4);
  float*    logit = (float*)take((size_t)MQ * 128 * 4);
  float*    val   = (float*)take((size_t)MS * Dm * 4);
  float*    acc   = (float*)take((size_t)MQ * Dm * 4);
  _Float16* acch  = (_Float16*)take((size_t)MQ * Dm * 2);
  float*    proj  = (float*)take((size_t)MQ * Dm * 4);
  float*    t1    = (float*)take((size_t)MQ * Dm * 4);
  _Float16* t1h   = (_Float16*)take((size_t)MQ * Dm * 2);
  float*    qkv   = (float*)take((size_t)MQ * 768 * 4);
  _Float16* qah   = (_Float16*)take((size_t)Bq * NH * LQ * HD * 2);
  _Float16* kah   = (_Float16*)take((size_t)Bq * NH * LQ * HD * 2);
  _Float16* vTh   = (_Float16*)take((size_t)Bq * NH * HD * LQ * 2);
  float*    ao    = (float*)take((size_t)MQ * Dm * 4);
  _Float16* aoh   = (_Float16*)take((size_t)MQ * Dm * 2);
  float*    mo    = (float*)take((size_t)MQ * Dm * 4);
  float*    t2    = (float*)take((size_t)MQ * Dm * 4);
  _Float16* t2h   = (_Float16*)take((size_t)MQ * Dm * 2);
  float*    f1    = (float*)take((size_t)MQ * FF * 4);
  _Float16* f1h   = (_Float16*)take((size_t)MQ * FF * 2);
  float*    f2    = (float*)take((size_t)MQ * Dm * 4);
  (void)ws_size; (void)in_sizes; (void)n_in; (void)out_size;

  auto cvt = [&](const float* s, _Float16* d, long long n) {
    k_cvt<<<dim3((unsigned)((n + 255) / 256)), dim3(256), 0, stream>>>(s, d, n);
  };
  // K==256 GEMMs go through the TDM (tensor_load_to_lds) path.
  auto gemm = [&](const _Float16* A, int lda, const _Float16* Wm, int ldw,
                  const float* bias, float* C, int ldc, int M, int N, int K,
                  int relu) {
    if (K == 256) {
      k_gemm_tdm<<<dim3(M / 16, N / 64), dim3(32), 0, stream>>>(
          A, lda, Wm, ldw, bias, C, ldc, relu);
    } else {
      k_gemm<<<dim3(M / 16, N / 64), dim3(32), 0, stream>>>(
          A, lda, Wm, ldw, bias, C, ldc, K, relu);
    }
  };

  // 1. q = tgt + query_pos  (f16) ;  src -> f16 ; weights -> f16
  k_add_cvt<<<dim3((MQ * Dm + 255) / 256), dim3(256), 0, stream>>>(
      tgt, qpos, qh, (long long)MQ * Dm);
  cvt(src, srch, (long long)MS * Dm);
  cvt(W_off, whoff, 256 * 256);  cvt(W_att, whatt, 128 * 256);
  cvt(W_val, whval, 256 * 256);  cvt(W_out, whout, 256 * 256);
  cvt(Wproj, whprj, 768 * 256);  cvt(Wmha, whmha, 256 * 256);
  cvt(W1, whw1, 1024 * 256);     cvt(W2, whw2, 256 * 1024);

  // 2. MSDA projections (WMMA GEMMs, TDM-staged A tiles)
  gemm(qh, 256, whoff, 256, b_off, offp, 256, MQ, 256, 256, 0);
  gemm(qh, 256, whatt, 256, b_att, logit, 128, MQ, 128, 256, 0);
  gemm(srch, 256, whval, 256, b_val, val, 256, MS, 256, 256, 0);

  // 3. Deformable sampling + attention-weight softmax
  k_msda<<<dim3(MQ * NH * 32 / 256), dim3(256), 0, stream>>>(
      offp, logit, ref, val, acc, acch);

  // 4. MSDA output projection, residual + LN1
  gemm(acch, 256, whout, 256, b_out, proj, 256, MQ, 256, 256, 0);
  k_ln_res<<<dim3(MQ / 8), dim3(256), 0, stream>>>(proj, tgt, ln1g, ln1b, t1, t1h, MQ);

  // 5. Self-attention: QKV GEMM, split, flash attention
  gemm(t1h, 256, whprj, 256, bproj, qkv, 768, MQ, 768, 256, 0);
  k_qkv_split<<<dim3((MQ * Dm + 255) / 256), dim3(256), 0, stream>>>(qkv, qah, kah, vTh);
  k_flash<<<dim3(LQ / 16, Bq * NH), dim3(32), 0, stream>>>(qah, kah, vTh, ao, aoh);

  // 6. MHA output projection, residual + LN2
  gemm(aoh, 256, whmha, 256, bmha, mo, 256, MQ, 256, 256, 0);
  k_ln_res<<<dim3(MQ / 8), dim3(256), 0, stream>>>(mo, t1, ln2g, ln2b, t2, t2h, MQ);

  // 7. FFN (ReLU GEMM -> GEMM), residual + LN3 -> d_out
  gemm(t2h, 256, whw1, 256, b1, f1, 1024, MQ, 1024, 256, 1);
  cvt(f1, f1h, (long long)MQ * FF);
  gemm(f1h, 1024, whw2, 1024, b2, f2, 256, MQ, 256, 1024, 0);
  k_ln_res<<<dim3(MQ / 8), dim3(256), 0, stream>>>(f2, t2, ln3g, ln3b, out,
                                                   (_Float16*)nullptr, MQ);
}
